// GPT_66468913873618
// MI455X (gfx1250) — matrix operand order repaired and checked by reading, not compiled, not run on hardware
//
#include <hip/hip_runtime.h>
#include <hip/hip_bf16.h>
#include <cstdint>

typedef __attribute__((ext_vector_type(16))) __bf16 v16bf;
typedef __attribute__((ext_vector_type(8)))  float  v8f;

#define GPT_L  8
#define GPT_H  16
#define GPT_D  1024
#define GPT_DH 64
#define GPT_FF 4096
#define GPT_V  32000
#define GPT_B  2
#define GPT_S  1024
#define GPT_M  (GPT_B * GPT_S)   // 2048 token rows

// ---------------------------------------------------------------------------
// pack two floats -> packed bf16x2 (round to nearest even), lo in bits[15:0]
// ---------------------------------------------------------------------------
__device__ __forceinline__ unsigned int pack2bf(float a, float b) {
  unsigned int ua = __float_as_uint(a);
  unsigned int ub = __float_as_uint(b);
  ua += 0x7FFFu + ((ua >> 16) & 1u);
  ub += 0x7FFFu + ((ub >> 16) & 1u);
  return (ua >> 16) | (ub & 0xFFFF0000u);
}

// ---------------------------------------------------------------------------
// Embedding: h[row, :] = tok_emb[x[row], :] + pos_emb[row % S, :]
// ---------------------------------------------------------------------------
__global__ __launch_bounds__(256) void gpt_embed(const int* __restrict__ x,
                                                 const float* __restrict__ tok,
                                                 const float* __restrict__ pos,
                                                 float* __restrict__ h) {
  const int row  = blockIdx.x;
  const int tid  = threadIdx.x;
  const int spos = row & (GPT_S - 1);
  const int t    = x[row];
  const size_t c = (size_t)tid * 4;
  float4 a = *(const float4*)(tok + (size_t)t * GPT_D + c);
  float4 b = *(const float4*)(pos + (size_t)spos * GPT_D + c);
  float4 o;
  o.x = a.x + b.x; o.y = a.y + b.y; o.z = a.z + b.z; o.w = a.w + b.w;
  *(float4*)(h + (size_t)row * GPT_D + c) = o;
}

// ---------------------------------------------------------------------------
// Tile staging: fp32 global -> bf16 LDS, pre-swizzled into CDNA5 WMMA
// fragment order. All global loads are issued before any conversion so they
// can stay in flight under the WMMAs of the current tile (double buffering).
//   A tile: 128 rows x 32 k   (fragment: lane=(m&15)+16*((k>>3)&1),
//                              elem j=((k>>4)<<3)|(k&7))
//   B tile: 32 k x 128 cols   (fragment: lane=(n&15)+16*((k>>3)&1),
//                              elem j=((k>>4)<<3)|(k&7))
// B is staged as a per-thread 4x4 transpose: 4 coalesced float4 row loads,
// then one 8-byte packed LDS store per column (no b16 scatter).
// ---------------------------------------------------------------------------
template <int WMODE>
__device__ __forceinline__ void gpt_stage(const float* __restrict__ A,
                                          const float* __restrict__ W,
                                          int N, int K, int m0, int n0, int k0,
                                          int tid,
                                          unsigned int* __restrict__ sA,
                                          unsigned int* __restrict__ sB) {
  // ---- issue all global loads first ----
  float4 a[4];
#pragma unroll
  for (int p = 0; p < 4; ++p) {
    const int flat4 = p * 256 + tid;      // 1024 float4 = 4096 elements
    const int row   = flat4 >> 3;         // 8 float4 per 32-wide k row
    const int kk    = (flat4 & 7) << 2;
    a[p] = *(const float4*)(A + (size_t)(m0 + row) * K + k0 + kk);
  }
  const int nq  = (tid & 31) << 2;        // 4 consecutive n
  const int kq  = (tid >> 5) << 2;        // 4 consecutive k
  const int ngb = n0 + nq;
  float4 b[4];
#pragma unroll
  for (int r = 0; r < 4; ++r) {
    const int kg = k0 + kq + r;
    const float* wp = (WMODE == 1)
        ? W + (size_t)(ngb >> 6) * ((size_t)K * 64) + (size_t)kg * 64 + (ngb & 63)
        : W + (size_t)kg * N + ngb;
    b[r] = *(const float4*)wp;
  }

  // ---- convert + packed 8-byte LDS stores (A) ----
#pragma unroll
  for (int p = 0; p < 4; ++p) {
    const int flat4 = p * 256 + tid;
    const int row   = flat4 >> 3;
    const int kk    = (flat4 & 7) << 2;
    const int lanep = ((row >> 4) * 32) + (row & 15) + (((kk >> 3) & 1) << 4);
    const int jb    = ((kk >> 4) << 3) | (kk & 7);     // multiple of 4
    uint2 pv;
    pv.x = pack2bf(a[p].x, a[p].y);
    pv.y = pack2bf(a[p].z, a[p].w);
    *(uint2*)(sA + ((lanep * 16 + jb) >> 1)) = pv;
  }

  // ---- convert + packed 8-byte LDS stores (B, 4x4 transpose) ----
  const int jb    = ((kq >> 4) << 3) | (kq & 7);        // multiple of 4
  const int lhalf = ((kq >> 3) & 1) << 4;
#pragma unroll
  for (int c = 0; c < 4; ++c) {
    const int n2    = nq + c;
    const int lanep = ((n2 >> 4) * 32) + (n2 & 15) + lhalf;
    const float v0 = ((const float*)&b[0])[c];
    const float v1 = ((const float*)&b[1])[c];
    const float v2 = ((const float*)&b[2])[c];
    const float v3 = ((const float*)&b[3])[c];
    uint2 pv;
    pv.x = pack2bf(v0, v1);
    pv.y = pack2bf(v2, v3);
    *(uint2*)(sB + ((lanep * 16 + jb) >> 1)) = pv;
  }
}

// ---------------------------------------------------------------------------
// WMMA GEMM:  C[M,N] = act(A[M,K] @ W + bias)
//   WMODE 0: W row-major [K,N];  WMODE 1: W head-blocked [H,K,64] as [K,H*64]
//   ACT 1: ReLU
// BM=BN=128, BK=32, double-buffered LDS (2 x 16KB). 256 threads = 8 waves in
// a 4x2 grid; each wave owns a 32x64 sub-tile = 2x4 v8f accumulators and
// issues 8 v_wmma_f32_16x16x32_bf16 per k-tile while the next tile streams in.
// ---------------------------------------------------------------------------
template <int WMODE, int ACT>
__global__ __launch_bounds__(256) void gpt_gemm(const float* __restrict__ A,
                                                const float* __restrict__ W,
                                                const float* __restrict__ bias,
                                                float* __restrict__ C,
                                                int M, int N, int K) {
  __shared__ __align__(32) unsigned int sA[2][128 * 32 / 2];
  __shared__ __align__(32) unsigned int sB[2][128 * 32 / 2];

  const int tid  = threadIdx.x;
  const int lane = tid & 31;
  const int wave = tid >> 5;
  const int wr   = wave >> 1;  // 0..3 : wave row (32 M-rows each)
  const int wc   = wave & 1;   // 0..1 : wave col (64 N-cols each)
  const int m0   = blockIdx.y * 128;
  const int n0   = blockIdx.x * 128;

  v8f zero = {};
  v8f acc[2][4];
#pragma unroll
  for (int i = 0; i < 2; ++i)
#pragma unroll
    for (int j = 0; j < 4; ++j) acc[i][j] = zero;

  const int KT = K >> 5;
  gpt_stage<WMODE>(A, W, N, K, m0, n0, 0, tid, sA[0], sB[0]);

  for (int kt = 0; kt < KT; ++kt) {
    __syncthreads();
    const int cur = kt & 1;
    if (kt + 1 < KT)
      gpt_stage<WMODE>(A, W, N, K, m0, n0, (kt + 1) << 5, tid,
                       sA[cur ^ 1], sB[cur ^ 1]);

    const unsigned short* bA = (const unsigned short*)sA[cur];
    const unsigned short* bB = (const unsigned short*)sB[cur];
    v16bf afrag[2], bfrag[4];
#pragma unroll
    for (int i = 0; i < 2; ++i)
      afrag[i] = *(const v16bf*)(bA + (((wr * 2 + i) * 32 + lane) << 4));
#pragma unroll
    for (int j = 0; j < 4; ++j)
      bfrag[j] = *(const v16bf*)(bB + (((wc * 4 + j) * 32 + lane) << 4));
#pragma unroll
    for (int i = 0; i < 2; ++i)
#pragma unroll
      for (int j = 0; j < 4; ++j)
        acc[i][j] = __builtin_amdgcn_wmma_f32_16x16x32_bf16(
            false, afrag[i], false, bfrag[j], (short)0, acc[i][j], false, false);
  }

  // ---- epilogue: C/D layout VGPR r -> M = r + 8*(lane>>4), N = lane&15 ----
  const int col_in = lane & 15;
  const int half   = lane >> 4;
#pragma unroll
  for (int i = 0; i < 2; ++i) {
#pragma unroll
    for (int j = 0; j < 4; ++j) {
      const int col = n0 + (wc * 4 + j) * 16 + col_in;
      const float bv = bias[col];
#pragma unroll
      for (int r = 0; r < 8; ++r) {
        const int rowm = m0 + (wr * 2 + i) * 16 + r + half * 8;
        float v = acc[i][j][r] + bv;
        if (ACT == 1) v = fmaxf(v, 0.0f);
        C[(size_t)rowm * N + col] = v;
      }
    }
  }
}

// ---------------------------------------------------------------------------
// Causal attention, online softmax. One wave per (b, h, s) query row.
// q/k/v layout: [B*S, H*64] row-major. DH=64 -> float2 per lane.
// ---------------------------------------------------------------------------
__global__ __launch_bounds__(256) void gpt_attn(const float* __restrict__ q,
                                                const float* __restrict__ k,
                                                const float* __restrict__ v,
                                                float* __restrict__ o) {
  const int gwid = blockIdx.x * 8 + (threadIdx.x >> 5);
  const int lane = threadIdx.x & 31;
  const int b  = gwid >> 14;            // / (H*S) = /16384
  const int rm = gwid & 16383;
  const int hh = rm >> 10;              // / S
  const int s  = rm & (GPT_S - 1);

  const size_t base = (size_t)b * GPT_S * GPT_D + (size_t)hh * GPT_DH + lane * 2;
  const float2 qv = *(const float2*)(q + base + (size_t)s * GPT_D);
  const float scale = 0.125f;           // 1/sqrt(64)

  float m = -3.402823466e38f, lsum = 0.0f;
  float ax = 0.0f, ay = 0.0f;
  for (int t = 0; t <= s; ++t) {
    const float2 kv = *(const float2*)(k + base + (size_t)t * GPT_D);
    float part = qv.x * kv.x + qv.y * kv.y;
#pragma unroll
    for (int off = 16; off > 0; off >>= 1) part += __shfl_xor(part, off, 32);
    const float sc = part * scale;
    const float nm = fmaxf(m, sc);
    const float corr = __expf(m - nm);
    const float p    = __expf(sc - nm);
    lsum = lsum * corr + p;
    const float2 vv = *(const float2*)(v + base + (size_t)t * GPT_D);
    ax = ax * corr + p * vv.x;
    ay = ay * corr + p * vv.y;
    m = nm;
  }
  const float inv = 1.0f / lsum;
  float2 res; res.x = ax * inv; res.y = ay * inv;
  *(float2*)(o + base + (size_t)s * GPT_D) = res;
}

// ---------------------------------------------------------------------------
// Fused residual + LayerNorm: h = LN(r + h) * g + b. One block per row.
// ---------------------------------------------------------------------------
__global__ __launch_bounds__(256) void gpt_add_ln(const float* __restrict__ r,
                                                  float* __restrict__ h,
                                                  const float* __restrict__ g,
                                                  const float* __restrict__ bb) {
  __shared__ float red[256];
  const int row = blockIdx.x, tid = threadIdx.x;
  const size_t base = (size_t)row * GPT_D + tid * 4;
  const float4 rv = *(const float4*)(r + base);
  const float4 hv = *(const float4*)(h + base);
  float x0 = rv.x + hv.x, x1 = rv.y + hv.y, x2 = rv.z + hv.z, x3 = rv.w + hv.w;

  red[tid] = x0 + x1 + x2 + x3;
  __syncthreads();
  for (int st = 128; st > 0; st >>= 1) { if (tid < st) red[tid] += red[tid + st]; __syncthreads(); }
  const float mu = red[0] * (1.0f / GPT_D);
  __syncthreads();

  const float d0 = x0 - mu, d1 = x1 - mu, d2 = x2 - mu, d3 = x3 - mu;
  red[tid] = d0 * d0 + d1 * d1 + d2 * d2 + d3 * d3;
  __syncthreads();
  for (int st = 128; st > 0; st >>= 1) { if (tid < st) red[tid] += red[tid + st]; __syncthreads(); }
  const float rstd = rsqrtf(red[0] * (1.0f / GPT_D) + 1e-5f);

  const int c = tid * 4;
  float4 o;
  o.x = d0 * rstd * g[c + 0] + bb[c + 0];
  o.y = d1 * rstd * g[c + 1] + bb[c + 1];
  o.z = d2 * rstd * g[c + 2] + bb[c + 2];
  o.w = d3 * rstd * g[c + 3] + bb[c + 3];
  *(float4*)(h + base) = o;
}

// ---------------------------------------------------------------------------
// In-place log-softmax over V=32000. One block per row.
// ---------------------------------------------------------------------------
__global__ __launch_bounds__(256) void gpt_logsoftmax(float* __restrict__ out) {
  __shared__ float red[256];
  const int row = blockIdx.x, tid = threadIdx.x;
  float* p = out + (size_t)row * GPT_V;

  float mx = -3.402823466e38f;
  for (int i = tid; i < GPT_V; i += 256) mx = fmaxf(mx, p[i]);
  red[tid] = mx;
  __syncthreads();
  for (int st = 128; st > 0; st >>= 1) { if (tid < st) red[tid] = fmaxf(red[tid], red[tid + st]); __syncthreads(); }
  mx = red[0];
  __syncthreads();

  float s = 0.0f;
  for (int i = tid; i < GPT_V; i += 256) s += __expf(p[i] - mx);
  red[tid] = s;
  __syncthreads();
  for (int st = 128; st > 0; st >>= 1) { if (tid < st) red[tid] += red[tid + st]; __syncthreads(); }
  const float lse = mx + logf(red[0]);
  __syncthreads();

  for (int i = tid; i < GPT_V; i += 256) p[i] = p[i] - lse;
}

// ---------------------------------------------------------------------------
// Orchestration
// ---------------------------------------------------------------------------
extern "C" void kernel_launch(void* const* d_in, const int* in_sizes, int n_in,
                              void* d_out, int out_size, void* d_ws, size_t ws_size,
                              hipStream_t stream) {
  (void)in_sizes; (void)n_in; (void)out_size; (void)ws_size;

  const int*   x       = (const int*)  d_in[0];
  const float* tok_emb = (const float*)d_in[1];
  const float* pos_emb = (const float*)d_in[2];
  const float* wq      = (const float*)d_in[3];
  const float* bq      = (const float*)d_in[4];
  const float* wk      = (const float*)d_in[5];
  const float* bk      = (const float*)d_in[6];
  const float* wv      = (const float*)d_in[7];
  const float* bv      = (const float*)d_in[8];
  const float* wo      = (const float*)d_in[9];
  const float* bo      = (const float*)d_in[10];
  const float* ln1_g   = (const float*)d_in[11];
  const float* ln1_b   = (const float*)d_in[12];
  const float* w1      = (const float*)d_in[13];
  const float* b1      = (const float*)d_in[14];
  const float* w2      = (const float*)d_in[15];
  const float* b2      = (const float*)d_in[16];
  const float* ln2_g   = (const float*)d_in[17];
  const float* ln2_b   = (const float*)d_in[18];
  const float* w_out   = (const float*)d_in[19];
  const float* b_out   = (const float*)d_in[20];
  float* out = (float*)d_out;

  // workspace carve-out (floats): h,q,k,v,o (2M each) + ff (8.39M) ~= 76 MB
  float* h  = (float*)d_ws;
  float* q  = h  + (size_t)GPT_M * GPT_D;
  float* kb = q  + (size_t)GPT_M * GPT_D;
  float* vb = kb + (size_t)GPT_M * GPT_D;
  float* o  = vb + (size_t)GPT_M * GPT_D;
  float* ff = o  + (size_t)GPT_M * GPT_D;

  gpt_embed<<<GPT_M, 256, 0, stream>>>(x, tok_emb, pos_emb, h);

  const dim3 gD (GPT_D  / 128, GPT_M / 128);   // N=1024
  const dim3 gFF(GPT_FF / 128, GPT_M / 128);   // N=4096
  const dim3 gV (GPT_V  / 128, GPT_M / 128);   // N=32000

  for (int l = 0; l < GPT_L; ++l) {
    const size_t oDD = (size_t)l * GPT_D * GPT_D;
    const size_t oD  = (size_t)l * GPT_D;
    const size_t oDF = (size_t)l * GPT_D * GPT_FF;
    const size_t oF  = (size_t)l * GPT_FF;

    // QKV projections (head-blocked weights)
    gpt_gemm<1, 0><<<gD, 256, 0, stream>>>(h, wq + oDD, bq + oD, q,  GPT_M, GPT_D, GPT_D);
    gpt_gemm<1, 0><<<gD, 256, 0, stream>>>(h, wk + oDD, bk + oD, kb, GPT_M, GPT_D, GPT_D);
    gpt_gemm<1, 0><<<gD, 256, 0, stream>>>(h, wv + oDD, bv + oD, vb, GPT_M, GPT_D, GPT_D);

    // causal attention
    gpt_attn<<<(GPT_B * GPT_H * GPT_S) / 8, 256, 0, stream>>>(q, kb, vb, o);

    // output projection -> tmp (reuse q), then residual+LN1
    gpt_gemm<0, 0><<<gD, 256, 0, stream>>>(o, wo + oDD, bo + oD, q, GPT_M, GPT_D, GPT_D);
    gpt_add_ln<<<GPT_M, 256, 0, stream>>>(q, h, ln1_g + oD, ln1_b + oD);

    // MLP: relu(h @ w1 + b1) @ w2 + b2, then residual+LN2
    gpt_gemm<0, 1><<<gFF, 256, 0, stream>>>(h,  w1 + oDF, b1 + oF, ff, GPT_M, GPT_FF, GPT_D);
    gpt_gemm<0, 0><<<gD,  256, 0, stream>>>(ff, w2 + oDF, b2 + oD, q,  GPT_M, GPT_D, GPT_FF);
    gpt_add_ln<<<GPT_M, 256, 0, stream>>>(q, h, ln2_g + oD, ln2_b + oD);
  }

  // logits straight into d_out, then in-place log-softmax
  gpt_gemm<0, 0><<<gV, 256, 0, stream>>>(h, w_out, b_out, out, GPT_M, GPT_V, GPT_D);
  gpt_logsoftmax<<<GPT_M, 256, 0, stream>>>(out);
}
